// RNN_MPN_25348896981720
// MI455X (gfx1250) — matrix operand order repaired and checked by reading, not compiled, not run on hardware
//
#include <hip/hip_runtime.h>
#include <hip/hip_bf16.h>

// ---------------------------------------------------------------------------
// RNN message passing (3-hop edge LSTM + scatter-sum) for MI455X / gfx1250.
// bf16 WMMA (v_wmma_f32_16x16x32_bf16), f32 accumulate. B matrices packed into
// WMMA fragment order and held in VGPRs. A tiles staged in LDS: gathered
// segments (node features, hidden state) via async global->LDS bf16 copies,
// streamed segment (edge_rep, HBM-bound) via f32 load + cvt_pk_bf16.
// ---------------------------------------------------------------------------

#define N_NODES 50000
#define N_EDGES 640000
#define RDIM    128
#define HDIM    128
#define K1      384   // 2R + H (gate GEMM K)
#define KC1     12    // 384 / 32
#define NT1     32    // 512 / 16 gate-column tiles
#define K2      256   // R + H (output GEMM K)
#define KC2     8
#define NT2     8     // 128 / 16 output-column tiles

typedef __attribute__((ext_vector_type(16))) __bf16 v16bf;
typedef __attribute__((ext_vector_type(8)))  __bf16 v8bf;
typedef __attribute__((ext_vector_type(4)))  __bf16 v4bf;
typedef __attribute__((ext_vector_type(8)))  float  v8f;
typedef __attribute__((ext_vector_type(4)))  float  v4f;

// LDS row padding (bf16 units) so 16 rows rotate across bank groups.
#define APAD1 392   // 384 + 8  -> row stride 196 dwords (rotate 4 banks/row)
#define APAD2 264   // 256 + 8  -> row stride 132 dwords (rotate 4 banks/row)

// ---- CDNA5 async global->LDS (ASYNCcnt) ----------------------------------
__device__ __forceinline__ void async_ld16(void* lds_ptr, const void* gptr) {
  unsigned int  ldsoff = (unsigned int)(unsigned long long)(uintptr_t)lds_ptr;
  unsigned long long ga = (unsigned long long)(uintptr_t)gptr;
  asm volatile("global_load_async_to_lds_b128 %0, %1, off"
               :
               : "v"(ldsoff), "v"(ga)
               : "memory");
}
__device__ __forceinline__ void wait_async0() {
  asm volatile("s_wait_asynccnt 0x0" ::: "memory");
}

// ---------------------------------------------------------------------------
// Pack [W_ih | W_hh]^T (384 x 512) into WMMA B-fragment order, bf16.
// Per 32x16 K-chunk: lane l<16 holds column N=l, K=0..15 contiguous;
// lane 16+l holds column N=l, K=16..31.
//   flat = ((ntile*KC + kc)*32 + lane)*16 + e,  k = kc*32 + (lane>>4)*16 + e
// ---------------------------------------------------------------------------
__global__ __launch_bounds__(256) void pack_w_kernel(
    const float* __restrict__ Wih, const float* __restrict__ Whh,
    __bf16* __restrict__ Wpack) {
  const int total = NT1 * KC1 * 32 * 16;
  for (int idx = blockIdx.x * blockDim.x + threadIdx.x; idx < total;
       idx += gridDim.x * blockDim.x) {
    int e    = idx & 15;
    int lane = (idx >> 4) & 31;
    int rest = idx >> 9;
    int kc   = rest % KC1;
    int nt   = rest / KC1;
    int n = nt * 16 + (lane & 15);
    int k = kc * 32 + (lane >> 4) * 16 + e;
    float v = (k < 256) ? Wih[n * 256 + k] : Whh[n * 128 + (k - 256)];
    Wpack[idx] = (__bf16)v;
  }
}

__global__ __launch_bounds__(256) void pack_wout_kernel(
    const float* __restrict__ Wout, __bf16* __restrict__ Wpack) {
  const int total = NT2 * KC2 * 32 * 16;
  for (int idx = blockIdx.x * blockDim.x + threadIdx.x; idx < total;
       idx += gridDim.x * blockDim.x) {
    int e    = idx & 15;
    int lane = (idx >> 4) & 31;
    int rest = idx >> 9;
    int kc   = rest % KC2;
    int nt   = rest / KC2;
    int n = nt * 16 + (lane & 15);
    int k = kc * 32 + (lane >> 4) * 16 + e;
    Wpack[idx] = (__bf16)Wout[n * 256 + k];
  }
}

// node_rep [N,128] f32 -> bf16 mirror (done once; L2-resident gather source)
__global__ __launch_bounds__(256) void node2bf_kernel(
    const float* __restrict__ node_rep, __bf16* __restrict__ nodeBf) {
  const int total = N_NODES * 32;  // float4 groups
  for (int q = blockIdx.x * blockDim.x + threadIdx.x; q < total;
       q += gridDim.x * blockDim.x) {
    v4f v = *(const v4f*)(node_rep + (size_t)q * 4);
    v4bf o;
    o[0] = (__bf16)v[0]; o[1] = (__bf16)v[1];
    o[2] = (__bf16)v[2]; o[3] = (__bf16)v[3];
    *(v4bf*)(nodeBf + (size_t)q * 4) = o;
  }
}

// hidden half of state [N,2,H] f32 -> [N,H] bf16 mirror (after each hop)
__global__ __launch_bounds__(256) void hid2bf_kernel(
    const float* __restrict__ state, __bf16* __restrict__ hidBf) {
  const int total = N_NODES * 32;  // float4 groups over N*128
  for (int q = blockIdx.x * blockDim.x + threadIdx.x; q < total;
       q += gridDim.x * blockDim.x) {
    int n = q >> 5;
    int u = (q & 31) * 4;
    v4f v = *(const v4f*)(state + (size_t)n * 256 + u);
    v4bf o;
    o[0] = (__bf16)v[0]; o[1] = (__bf16)v[1];
    o[2] = (__bf16)v[2]; o[3] = (__bf16)v[3];
    *(v4bf*)(hidBf + (size_t)n * 128 + u) = o;
  }
}

__device__ __forceinline__ float fast_sigmoid(float x) {
  return 1.0f / (1.0f + __expf(-x));
}
__device__ __forceinline__ float fast_tanh(float x) {
  return 2.0f / (1.0f + __expf(-2.0f * x)) - 1.0f;
}

// ---------------------------------------------------------------------------
// One message-passing hop.  grid = (E/256, 4), block = 256 (8 waves).
// blockIdx.y selects 32 hidden units; block computes i,f,g,o columns for
// those units (8 n-tiles, one per wave) so the LSTM completes in-block.
// ---------------------------------------------------------------------------
__global__ __launch_bounds__(256) void hop_kernel(
    const float* __restrict__ state_in,    // [N,2,H] f32 (cell gather)
    const __bf16* __restrict__ hidBf,      // [N,H] bf16 hidden mirror
    float* __restrict__ state_out,         // [N,2,H] pre-zeroed, atomic acc
    const float* __restrict__ edge_rep,    // [E,R] f32 (HBM stream)
    const __bf16* __restrict__ nodeBf,     // [N,R] bf16 mirror
    const int* __restrict__ src,
    const int* __restrict__ dst,
    const __bf16* __restrict__ Wpack,
    const float* __restrict__ b_ih,
    const float* __restrict__ b_hh) {
  __shared__ __bf16 sA[16 * APAD1];        // 16-edge x 384 A tile (bf16)
  __shared__ float  sC[16][4][33];         // C exchange: [edge][gate][unit32]
  __shared__ int    sSrc[16], sDst[16];
  __shared__ float  sBias[512];

  const int t    = threadIdx.x;
  const int lane = t & 31;
  const int wave = t >> 5;
  const int gate = wave >> 1;
  const int p    = wave & 1;
  const int ugrp = blockIdx.y;             // hidden units [32*ugrp, +32)
  const int ntile = gate * 8 + ugrp * 2 + p;
  const int hi = lane >> 4;
  const int m  = lane & 15;

  sBias[t]       = b_ih[t]       + b_hh[t];
  sBias[t + 256] = b_ih[t + 256] + b_hh[t + 256];

  // B fragments, all K=384 register-resident: 96 VGPRs.
  v16bf bfrag[KC1];
  {
    const __bf16* wp = Wpack + (size_t)ntile * KC1 * 512 + (size_t)lane * 16;
#pragma unroll
    for (int kc = 0; kc < KC1; ++kc)
      bfrag[kc] = *(const v16bf*)(wp + kc * 512);
  }

  const int ebase0 = blockIdx.x * 256;
  const int rstg  = t >> 4;    // staging row (0..15)
  const int sl    = t & 15;    // staging lane-in-row (16B granules)

  for (int rt = 0; rt < 16; ++rt) {
    const int eb = ebase0 + rt * 16;
    __syncthreads();  // previous iteration's LDS readers done
    if (t < 16) { sSrc[t] = src[eb + t]; sDst[t] = dst[eb + t]; }

    // --- async gather: node features (cols 0..127) + hidden (cols 256..383)
    {
      const int s = src[eb + rstg];
      async_ld16(&sA[rstg * APAD1 + sl * 8],
                 nodeBf + (size_t)s * 128 + sl * 8);
      async_ld16(&sA[rstg * APAD1 + 256 + sl * 8],
                 hidBf + (size_t)s * 128 + sl * 8);
    }
    // --- streamed edge features (cols 128..255): f32 -> bf16
#pragma unroll
    for (int i = 0; i < 2; ++i) {
      int q  = t + i * 256;            // 0..511 (16 rows x 32 float4)
      int r  = q >> 5;
      int cc = (q & 31) * 4;
      int e  = eb + r;
      v4f v = *(const v4f*)(edge_rep + (size_t)e * 128 + cc);
      if (rt < 15)
        __builtin_prefetch(edge_rep + (size_t)(e + 16) * 128 + cc, 0, 1);
      v4bf o;
      o[0] = (__bf16)v[0]; o[1] = (__bf16)v[1];
      o[2] = (__bf16)v[2]; o[3] = (__bf16)v[3];
      *(v4bf*)&sA[r * APAD1 + 128 + cc] = o;
    }
    wait_async0();
    __syncthreads();

    // --- GEMM: K=384 via 12 bf16 WMMAs, B in registers.
    v8f acc = {0.f, 0.f, 0.f, 0.f, 0.f, 0.f, 0.f, 0.f};
#pragma unroll
    for (int kc = 0; kc < KC1; ++kc) {
      const v8bf alo = *(const v8bf*)&sA[m * APAD1 + kc * 32 + hi * 16];
      const v8bf ahi = *(const v8bf*)&sA[m * APAD1 + kc * 32 + hi * 16 + 8];
      v16bf a = __builtin_shufflevector(alo, ahi, 0, 1, 2, 3, 4, 5, 6, 7,
                                        8, 9, 10, 11, 12, 13, 14, 15);
      acc = __builtin_amdgcn_wmma_f32_16x16x32_bf16(
          false, a, false, bfrag[kc], (short)0, acc, false, false);
    }

    // --- C tile -> LDS: element v of lane = (edge v+hi*8, unit p*16+m)
    {
      const int ul = p * 16 + m;
#pragma unroll
      for (int v = 0; v < 8; ++v) sC[v + hi * 8][gate][ul] = acc[v];
    }
    __syncthreads();

    // --- LSTM elementwise for 16 edges x 32 units, scatter-add
#pragma unroll
    for (int i2 = 0; i2 < 2; ++i2) {
      int it  = t + i2 * 256;          // 0..511
      int e   = it >> 5;
      int ulx = it & 31;
      int unit = ugrp * 32 + ulx;
      float iv = sC[e][0][ulx] + sBias[unit];
      float fv = sC[e][1][ulx] + sBias[128 + unit];
      float gv = sC[e][2][ulx] + sBias[256 + unit];
      float ov = sC[e][3][ulx] + sBias[384 + unit];
      float cprev = state_in[(size_t)sSrc[e] * 256 + 128 + unit];
      float c2 = fast_sigmoid(fv) * cprev + fast_sigmoid(iv) * fast_tanh(gv);
      float h2 = fast_sigmoid(ov) * fast_tanh(c2);
      float* op = state_out + (size_t)sDst[e] * 256 + unit;
      atomicAdd(op, h2);
      atomicAdd(op + 128, c2);
    }
  }
}

// ---------------------------------------------------------------------------
// Output: relu([node_rep | h] @ W_out^T + b_out).  grid = 625, block = 256.
// A tile fully async-staged from bf16 mirrors.
// ---------------------------------------------------------------------------
__global__ __launch_bounds__(256) void out_kernel(
    const __bf16* __restrict__ nodeBf,     // [N,R] bf16
    const __bf16* __restrict__ hidBf,      // [N,H] bf16
    const __bf16* __restrict__ WoutPack,
    const float* __restrict__ b_out,
    float* __restrict__ out) {             // [N,R] f32
  __shared__ __bf16 sB[16 * APAD2];

  const int t = threadIdx.x;
  const int lane = t & 31;
  const int wave = t >> 5;                 // == ntile
  const int hi = lane >> 4;
  const int m  = lane & 15;
  const int col = wave * 16 + m;
  const float bo = b_out[col];
  const int rstg = t >> 4;
  const int sl   = t & 15;

  v16bf bfrag[KC2];
  {
    const __bf16* wp = WoutPack + (size_t)wave * KC2 * 512 + (size_t)lane * 16;
#pragma unroll
    for (int kc = 0; kc < KC2; ++kc)
      bfrag[kc] = *(const v16bf*)(wp + kc * 512);
  }

  for (int rt = 0; rt < 5; ++rt) {
    const int rowbase = (blockIdx.x * 5 + rt) * 16;   // 3125 tiles exactly
    __syncthreads();
    {
      const int n = rowbase + rstg;
      async_ld16(&sB[rstg * APAD2 + sl * 8],
                 nodeBf + (size_t)n * 128 + sl * 8);
      async_ld16(&sB[rstg * APAD2 + 128 + sl * 8],
                 hidBf + (size_t)n * 128 + sl * 8);
    }
    wait_async0();
    __syncthreads();

    v8f acc = {0.f, 0.f, 0.f, 0.f, 0.f, 0.f, 0.f, 0.f};
#pragma unroll
    for (int kc = 0; kc < KC2; ++kc) {
      const v8bf alo = *(const v8bf*)&sB[m * APAD2 + kc * 32 + hi * 16];
      const v8bf ahi = *(const v8bf*)&sB[m * APAD2 + kc * 32 + hi * 16 + 8];
      v16bf a = __builtin_shufflevector(alo, ahi, 0, 1, 2, 3, 4, 5, 6, 7,
                                        8, 9, 10, 11, 12, 13, 14, 15);
      acc = __builtin_amdgcn_wmma_f32_16x16x32_bf16(
          false, a, false, bfrag[kc], (short)0, acc, false, false);
    }
#pragma unroll
    for (int v = 0; v < 8; ++v) {
      int grow = rowbase + v + hi * 8;
      out[(size_t)grow * 128 + col] = fmaxf(acc[v] + bo, 0.0f);
    }
  }
}

// ---------------------------------------------------------------------------
extern "C" void kernel_launch(void* const* d_in, const int* in_sizes, int n_in,
                              void* d_out, int out_size, void* d_ws,
                              size_t ws_size, hipStream_t stream) {
  const float* node_rep   = (const float*)d_in[0];
  const float* edge_rep   = (const float*)d_in[1];
  const float* init_state = (const float*)d_in[2];
  const float* W_ih       = (const float*)d_in[3];
  const float* W_hh       = (const float*)d_in[4];
  const float* b_ih       = (const float*)d_in[5];
  const float* b_hh       = (const float*)d_in[6];
  const float* W_out      = (const float*)d_in[7];
  const float* b_out      = (const float*)d_in[8];
  const int*   src        = (const int*)d_in[9];
  const int*   dst        = (const int*)d_in[10];
  float* out = (float*)d_out;

  const size_t S  = (size_t)N_NODES * 2 * HDIM * sizeof(float);  // 51.2 MB
  const size_t HB = (size_t)N_NODES * HDIM * sizeof(__bf16);     // 12.8 MB
  char* w = (char*)d_ws;
  float*  stateA   = (float*)w;
  float*  stateB   = (float*)(w + S);
  __bf16* Wpack    = (__bf16*)(w + 2 * S);                  // 393 KB
  __bf16* WoutPack = (__bf16*)(w + 2 * S + (512 << 10));    // 64 KB
  __bf16* nodeBf   = (__bf16*)(w + 2 * S + (1 << 20));
  __bf16* hidA     = (__bf16*)(w + 2 * S + (1 << 20) + HB);
  __bf16* hidB     = (__bf16*)(w + 2 * S + (1 << 20) + 2 * HB);

  pack_w_kernel<<<768, 256, 0, stream>>>(W_ih, W_hh, Wpack);
  pack_wout_kernel<<<128, 256, 0, stream>>>(W_out, WoutPack);
  node2bf_kernel<<<2048, 256, 0, stream>>>(node_rep, nodeBf);
  hid2bf_kernel<<<2048, 256, 0, stream>>>(init_state, hidA);

  dim3 hgrid(N_EDGES / 256, 4);

  hipMemsetAsync(stateB, 0, S, stream);
  hop_kernel<<<hgrid, 256, 0, stream>>>(init_state, hidA, stateB, edge_rep,
                                        nodeBf, src, dst, Wpack, b_ih, b_hh);
  hid2bf_kernel<<<2048, 256, 0, stream>>>(stateB, hidB);
  hipMemsetAsync(stateA, 0, S, stream);
  hop_kernel<<<hgrid, 256, 0, stream>>>(stateB, hidB, stateA, edge_rep,
                                        nodeBf, src, dst, Wpack, b_ih, b_hh);
  hid2bf_kernel<<<2048, 256, 0, stream>>>(stateA, hidA);
  hipMemsetAsync(stateB, 0, S, stream);
  hop_kernel<<<hgrid, 256, 0, stream>>>(stateA, hidA, stateB, edge_rep,
                                        nodeBf, src, dst, Wpack, b_ih, b_hh);
  hid2bf_kernel<<<2048, 256, 0, stream>>>(stateB, hidB);

  out_kernel<<<625, 256, 0, stream>>>(nodeBf, hidB, WoutPack, b_out, out);
}
